// Attention_54872502174186
// MI455X (gfx1250) — compile-verified, matrix operand order checked by reading, not run on hardware
//
#include <hip/hip_runtime.h>
#include <hip/hip_bf16.h>
#include <math.h>

typedef __attribute__((ext_vector_type(16))) _Float16 v16h;
typedef __attribute__((ext_vector_type(8)))  float    v8f;

#define NEG_SLOPE 0.01f
#define NUM_EDGES 20000
#define DIM 128

// ---------------------------------------------------------------------------
// Kernel 1: per-row MLP + attention logit via WMMA.
// One wave (32 lanes) handles a 16-row tile of the M gathered rows.
// h = leaky_relu(X[node] @ W1 + b1);  w = h @ W2 + b2
//
// A fragment (16x32 f16, ISA 7.12.2): lanes 0-15 <-> rows, per-lane halfs are
// K = kb..kb+7, kb+16..kb+23 with kb = Kbase + (lane>=16 ? 8 : 0).
// B fragments are PRE-SWIZZLED into LDS in exactly this per-lane order so the
// inner loop reads each 16-half fragment as one contiguous 32-byte DS load
// (2x ds_load_b128) instead of 16x ds_load_u16.
// C/D (16x16 f32): VGPR v, lanes 0-15 -> (M=v, N=lane); lanes 16-31 -> M=v+8.
// ---------------------------------------------------------------------------
__global__ __launch_bounds__(256) void mlp_logits_wmma(
    const float* __restrict__ X, const float* __restrict__ W1,
    const float* __restrict__ b1, const float* __restrict__ W2,
    const float* __restrict__ b2, const int* __restrict__ node_idx,
    float* __restrict__ wlogit, int M)
{
    // 32 fragments (nt*4+kk) x 32 lanes x 16 halfs = 16384 halfs = 32 KB
    __shared__ __align__(32) _Float16 sB[32 * 32 * 16];
    __shared__ float sb1[DIM];
    __shared__ float sW2[DIM];

    // Pre-swizzle W1 (row-major [k][n]) into per-lane B-fragment layout.
    for (int idx = threadIdx.x; idx < 32 * 32 * 16; idx += 256) {
        const int i    = idx & 15;         // element within lane chunk
        const int ln   = (idx >> 4) & 31;  // lane
        const int frag = idx >> 9;         // nt*4 + kk
        const int kk   = frag & 3;
        const int nt   = frag >> 2;
        const int col  = nt * 16 + (ln & 15);
        const int kb   = kk * 32 + ((ln >> 4) ? 8 : 0);
        const int k    = kb + (i < 8 ? i : 8 + i);   // i<8: kb+i ; else kb+16+(i-8)
        sB[idx] = (_Float16)W1[k * DIM + col];
    }
    if (threadIdx.x < DIM) {
        sb1[threadIdx.x] = b1[threadIdx.x];
        sW2[threadIdx.x] = W2[threadIdx.x];
    }
    __syncthreads();

    const int wave  = threadIdx.x >> 5;
    const int lane  = threadIdx.x & 31;
    const int tile  = blockIdx.x * 8 + wave;
    const int tiles = (M + 15) >> 4;
    if (tile >= tiles) return;

    const int row = lane & 15;
    const int hi8 = (lane >> 4) ? 8 : 0;
    int gr = tile * 16 + row;
    if (gr >= M) gr = M - 1;                 // clamp (M is a multiple of 16 anyway)
    const int node = node_idx[gr];
    const float* xr = X + (size_t)node * DIM;

    // Build all four K-step A fragments for this 16-row tile (f32 -> f16).
    v16h A[4];
#pragma unroll
    for (int kk = 0; kk < 4; ++kk) {
        const int kb = kk * 32 + hi8;
#pragma unroll
        for (int i = 0; i < 8; ++i) {
            A[kk][i]     = (_Float16)xr[kb + i];
            A[kk][8 + i] = (_Float16)xr[kb + 16 + i];
        }
    }

    float wacc[8];
#pragma unroll
    for (int v = 0; v < 8; ++v) wacc[v] = 0.f;

    const int col0 = lane & 15;
#pragma unroll
    for (int nt = 0; nt < 8; ++nt) {              // 8 N-tiles of 16 -> H = 128
        v8f c = {};
#pragma unroll
        for (int kk = 0; kk < 4; ++kk) {          // 4 K-steps of 32 -> d = 128
            const v16h B =
                *(const v16h*)&sB[(((nt << 2) + kk) * 32 + lane) * 16];
            c = __builtin_amdgcn_wmma_f32_16x16x32_f16(
                    /*neg_a=*/false, A[kk], /*neg_b=*/false, B,
                    /*c_mod=*/(short)0, c, /*reuse_a=*/false, /*reuse_b=*/false);
        }
        // bias + leaky_relu on f32 accumulator, then dot with W2 column coeff
        const int   col = nt * 16 + col0;
        const float bc  = sb1[col];
        const float w2c = sW2[col];
#pragma unroll
        for (int v = 0; v < 8; ++v) {
            float h = c[v] + bc;
            h = h > 0.f ? h : NEG_SLOPE * h;
            wacc[v] += h * w2c;
        }
    }

    // Reduce partial row-dots across the 16 lanes of each half-wave.
#pragma unroll
    for (int v = 0; v < 8; ++v) {
        float s = wacc[v];
        s += __shfl_xor(s, 1);
        s += __shfl_xor(s, 2);
        s += __shfl_xor(s, 4);
        s += __shfl_xor(s, 8);
        wacc[v] = s;
    }

    const float b2v = b2[0];
    if (lane == 0 || lane == 16) {
        const int rbase = tile * 16 + (lane ? 8 : 0);
#pragma unroll
        for (int v = 0; v < 8; ++v) {
            const int r = rbase + v;
            if (r < M) wlogit[r] = wacc[v] + b2v;
        }
    }
}

// ---------------------------------------------------------------------------
// Kernel 2: CSR offsets from sorted edge_ids via per-edge lower_bound.
// ---------------------------------------------------------------------------
__global__ void edge_offsets(const int* __restrict__ edge_ids, int M, int E,
                             int* __restrict__ start)
{
    const int e = blockIdx.x * blockDim.x + threadIdx.x;
    if (e > E) return;
    if (e == E) { start[E] = M; return; }
    int lo = 0, hi = M;
    while (lo < hi) {
        const int mid = (lo + hi) >> 1;
        if (edge_ids[mid] < e) lo = mid + 1; else hi = mid;
    }
    start[e] = lo;
}

// ---------------------------------------------------------------------------
// Kernel 3: per-edge segment softmax + weighted feature aggregation.
// One 128-thread block per hyperedge (avg 32 members). Thread t owns dim t.
// Next chunk's gathered X rows are prefetched while the current chunk is
// accumulated (X fits in L2; prefetch hides the L2 hit latency).
// ---------------------------------------------------------------------------
__global__ __launch_bounds__(128) void edge_softmax_agg(
    const float* __restrict__ X, const int* __restrict__ node_idx,
    const float* __restrict__ wlogit, const int* __restrict__ start,
    float* __restrict__ outZ, float* __restrict__ beta, int M)
{
    const int e   = blockIdx.x;
    const int s   = start[e];
    const int t   = start[e + 1];
    const int tid = threadIdx.x;

    __shared__ float red[DIM];
    __shared__ float sb[DIM];
    __shared__ int   sn[DIM];

    // segment max
    float mx = -INFINITY;
    for (int i = s + tid; i < t; i += DIM) mx = fmaxf(mx, wlogit[i]);
    red[tid] = mx; __syncthreads();
    for (int o = 64; o > 0; o >>= 1) {
        if (tid < o) red[tid] = fmaxf(red[tid], red[tid + o]);
        __syncthreads();
    }
    const float m = red[0];
    __syncthreads();

    // sum of exp; stash unnormalized ex into beta output
    float sm = 0.f;
    for (int i = s + tid; i < t; i += DIM) {
        const float ex = __expf(wlogit[i] - m);
        beta[i] = ex;
        sm += ex;
    }
    red[tid] = sm; __syncthreads();
    for (int o = 64; o > 0; o >>= 1) {
        if (tid < o) red[tid] += red[tid + o];
        __syncthreads();
    }
    const float denom = red[0];
    const float inv   = (t > s) ? 1.0f / denom : 0.0f;
    __syncthreads();

    // normalize beta in place and accumulate Z[dim] over LDS-staged chunks
    float acc = 0.f;
    for (int base = s; base < t; base += DIM) {
        const int i = base + tid;
        if (i < t) {
            const float b = beta[i] * inv;
            beta[i] = b;
            sb[tid] = b;
            sn[tid] = node_idx[i];
        }
        // prefetch next chunk's gathered rows (global_prefetch_b8)
        const int ipf = i + DIM;
        if (ipf < t)
            __builtin_prefetch(X + (size_t)node_idx[ipf] * DIM, 0, 0);
        __syncthreads();
        const int chunk = min(DIM, t - base);
        for (int j = 0; j < chunk; ++j)
            acc += sb[j] * X[(size_t)sn[j] * DIM + tid];
        __syncthreads();
    }

    const float z = acc > 0.f ? acc : NEG_SLOPE * acc;
    outZ[(size_t)e * DIM + tid] = tanhf(z);
}

// ---------------------------------------------------------------------------
// Host launcher. Input order: X, W1, b1, W2, b2, node_idx, edge_ids.
// Output: tanh(Z) [E*128] then beta [M], concatenated.
// ---------------------------------------------------------------------------
extern "C" void kernel_launch(void* const* d_in, const int* in_sizes, int n_in,
                              void* d_out, int out_size, void* d_ws, size_t ws_size,
                              hipStream_t stream)
{
    const float* X        = (const float*)d_in[0];
    const float* W1       = (const float*)d_in[1];
    const float* b1       = (const float*)d_in[2];
    const float* W2       = (const float*)d_in[3];
    const float* b2       = (const float*)d_in[4];
    const int*   node_idx = (const int*)d_in[5];
    const int*   edge_ids = (const int*)d_in[6];

    const int M = in_sizes[5];
    const int E = NUM_EDGES;

    float* wlogit = (float*)d_ws;                                    // M floats
    int*   start  = (int*)((char*)d_ws + (size_t)M * sizeof(float)); // E+1 ints

    float* outZ  = (float*)d_out;                   // E*128
    float* betaO = (float*)d_out + (size_t)E * DIM; // M

    const int tiles  = (M + 15) / 16;
    const int blocks = (tiles + 7) / 8;     // 8 waves (one tile each) per block
    mlp_logits_wmma<<<blocks, 256, 0, stream>>>(X, W1, b1, W2, b2, node_idx,
                                                wlogit, M);

    edge_offsets<<<(E + 1 + 255) / 256, 256, 0, stream>>>(edge_ids, M, E, start);

    edge_softmax_agg<<<E, DIM, 0, stream>>>(X, node_idx, wlogit, start,
                                            outZ, betaO, M);
}
